// GRUModel_53025666237056
// MI455X (gfx1250) — compile-verified, matrix operand order checked by reading, not run on hardware
//
#include <hip/hip_runtime.h>

#define Bsz 128
#define Tn  512
#define In  64
#define Hn  256
#define G3  768   // 3*H

typedef __attribute__((ext_vector_type(16))) __bf16 bf16x16;
typedef __attribute__((ext_vector_type(8)))  float  floatx8;

// Pack two f32 -> two bf16 (round-half-up) in one v_perm_b32 + 2 adds.
__device__ __forceinline__ unsigned pack2(float lo, float hi) {
  union { float f; unsigned u; } a, b;
  a.f = lo; b.f = hi;
  return __builtin_amdgcn_perm(b.u + 0x8000u, a.u + 0x8000u, 0x07060302u);
}
union BfFrag { unsigned u[8]; bf16x16 v; };

// ---------------------------------------------------------------------------
// Pre-swizzle a row-major f32 weight matrix W[N][K] into bf16 B-fragments.
// out index = ((nt*Kt + kt)*32 + lane)*8 + j  (each uint = 2 packed bf16)
// ---------------------------------------------------------------------------
__global__ void swizzle_weights(const float* __restrict__ W,
                                unsigned* __restrict__ out, int N, int K) {
  int idx = blockIdx.x * blockDim.x + threadIdx.x;
  int total = N * K / 2;
  if (idx >= total) return;
  int Kt  = K >> 5;
  int j    = idx & 7;
  int lane = (idx >> 3) & 31;
  int tile = idx >> 8;
  int kt = tile % Kt;
  int nt = tile / Kt;
  int n  = nt * 16 + (lane & 15);
  int hh = lane >> 4;
  int k0 = kt * 32 + (j < 4 ? hh * 8 + 2 * j : 16 + hh * 8 + 2 * (j - 4));
  out[idx] = pack2(W[n * K + k0], W[n * K + k0 + 1]);
}

// ---------------------------------------------------------------------------
// Parallel GEMM: C[M][768] = A[M][K] * Bswᵀ + bias.
// One block (8 waves) = one 16-row M-tile; each wave owns 6 of the 48
// N-tiles, so each packed A fragment feeds 6 back-to-back WMMAs.
// All indices 32-bit (everything < 2^31 elements).
// ---------------------------------------------------------------------------
__global__ __launch_bounds__(256) void gemm_bf16_wmma(
    const float* __restrict__ A, const unsigned* __restrict__ Bsw,
    const float* __restrict__ bias, float* __restrict__ C, int K) {
  const int lane = threadIdx.x & 31;
  const int Kt = K >> 5;
  const int ml = lane & 15, half = lane >> 4;
  const int mt = blockIdx.x;
  const int ng = threadIdx.x >> 5;          // 0..7 -> N-tiles ng*6 .. ng*6+5
  const float* arow = A + (unsigned)((mt * 16 + ml) * K);
  floatx8 acc[6] = {};
  for (int kt = 0; kt < Kt; ++kt) {
    int kbase = kt * 32;
    // A fragment: lane holds row M=ml; K pairs per documented 16-bit A layout
    const float4* pa = (const float4*)(arow + kbase + half * 8);
    const float4* pb = (const float4*)(arow + kbase + 16 + half * 8);
    float4 a0 = pa[0], a1 = pa[1], a2 = pb[0], a3 = pb[1];
    BfFrag af;
    af.u[0] = pack2(a0.x, a0.y); af.u[1] = pack2(a0.z, a0.w);
    af.u[2] = pack2(a1.x, a1.y); af.u[3] = pack2(a1.z, a1.w);
    af.u[4] = pack2(a2.x, a2.y); af.u[5] = pack2(a2.z, a2.w);
    af.u[6] = pack2(a3.x, a3.y); af.u[7] = pack2(a3.z, a3.w);
#pragma unroll
    for (int i = 0; i < 6; ++i) {
      int nt = ng * 6 + i;
      const bf16x16 bv =
          *(const bf16x16*)(Bsw + (unsigned)(((nt * Kt + kt) * 32 + lane) * 8));
      acc[i] = __builtin_amdgcn_wmma_f32_16x16x32_bf16(
          false, af.v, false, bv, (short)0, acc[i], false, false);
    }
  }
#pragma unroll
  for (int i = 0; i < 6; ++i) {
    int col = (ng * 6 + i) * 16 + ml;
    float bb = bias[col];
    int base = (mt * 16 + 8 * half) * G3 + col;   // documented C layout
#pragma unroll
    for (int j = 0; j < 8; ++j)
      C[(unsigned)(base + j * G3)] = acc[i][j] + bb;
  }
}

// ---------------------------------------------------------------------------
// Recurrent GRU layer. 8 blocks (16 batch rows each); h lives in LDS for all
// T steps. Per step: 8 waves x 6 N-tiles x 8 K-steps of
// v_wmma_f32_16x16x32_bf16 reading pre-packed bf16 A-fragments from LDS,
// then gates. The gate phase writes h_new BOTH as fp32 (for the z*h term)
// and directly as packed bf16 A-fragments for the next step's WMMAs.
// b_hh lives in registers (only 8 workgroups exist -> VGPRs are free);
// gx rows for the step are prefetched during the WMMA phase.
// ---------------------------------------------------------------------------
__global__ __launch_bounds__(256) void gru_recurrent(
    const float* __restrict__ gx,      // [B][T][768], already includes b_ih
    const unsigned* __restrict__ Wsw,  // swizzled W_hh (768x256 bf16)
    const float* __restrict__ bhh,     // [768]
    float* __restrict__ out_seq,       // [B][T][256] (if store_seq)
    float* __restrict__ out_last,      // [B][256]    (if !store_seq)
    int store_seq) {
  __shared__ __align__(32) float    hsh[16 * Hn];    // 16 KB fp32 state
  __shared__ __align__(32) float    ghs[16 * G3];    // 48 KB gh tile
  __shared__ __align__(32) unsigned aswz[8][32][8];  //  8 KB bf16 A-frags

  const int tid  = threadIdx.x;
  const int lane = tid & 31, w = tid >> 5;
  const int ml = lane & 15, half = lane >> 4;
  const int b0 = blockIdx.x * 16;

  for (int i = tid; i < 16 * Hn; i += 256) hsh[i] = 0.f;
  for (int i = tid; i < 8 * 32 * 8; i += 256) ((unsigned*)aswz)[i] = 0u;
  __syncthreads();

  const int row = tid >> 4;          // gate phase: row 0..15
  const int c0  = (tid & 15) * 16;   // gate phase: 16 contiguous columns

  // step-invariant per-thread bias registers
  float4 br4[4], bz4[4], bn4[4];
#pragma unroll
  for (int q = 0; q < 4; ++q) {
    br4[q] = ((const float4*)(bhh + c0))[q];
    bz4[q] = ((const float4*)(bhh + Hn + c0))[q];
    bn4[q] = ((const float4*)(bhh + 2 * Hn + c0))[q];
  }

  for (int t = 0; t < Tn; ++t) {
    const float* gxp = gx + (unsigned)(((b0 + row) * Tn + t) * G3);
    // prefetch this step's gate inputs while the WMMAs run
    __builtin_prefetch(gxp + c0, 0, 3);
    __builtin_prefetch(gxp + Hn + c0, 0, 3);
    __builtin_prefetch(gxp + 2 * Hn + c0, 0, 3);

    // ---- gh = h * W_hhᵀ -------------------------------------------------
    bf16x16 afr[8];
#pragma unroll
    for (int kt = 0; kt < 8; ++kt)
      afr[kt] = *(const bf16x16*)&aswz[kt][lane][0];
#pragma unroll
    for (int i = 0; i < 6; ++i) {
      int nt = w * 6 + i;
      floatx8 acc = {};
#pragma unroll
      for (int kt = 0; kt < 8; ++kt) {
        const bf16x16 bv =
            *(const bf16x16*)(Wsw + (unsigned)(((nt * 8 + kt) * 32 + lane) * 8));
        acc = __builtin_amdgcn_wmma_f32_16x16x32_bf16(
            false, afr[kt], false, bv, (short)0, acc, false, false);
      }
#pragma unroll
      for (int j = 0; j < 8; ++j)
        ghs[(j + 8 * half) * G3 + nt * 16 + ml] = acc[j];
    }
    __syncthreads();

    // ---- gates + state update (4 chunks of float4) ----------------------
#pragma unroll
    for (int q = 0; q < 4; ++q) {
      int c = c0 + 4 * q;
      float4 xr = ((const float4*)(gxp + c))[0];
      float4 xz = ((const float4*)(gxp + Hn + c))[0];
      float4 xn = ((const float4*)(gxp + 2 * Hn + c))[0];
      float4 sr = ((const float4*)(ghs + row * G3 + c))[0];
      float4 sz = ((const float4*)(ghs + row * G3 + Hn + c))[0];
      float4 sn = ((const float4*)(ghs + row * G3 + 2 * Hn + c))[0];
      float4 hc = ((const float4*)(hsh + row * Hn + c))[0];
      float hn4[4];
      const float rr[4] = {xr.x + sr.x + br4[q].x, xr.y + sr.y + br4[q].y,
                           xr.z + sr.z + br4[q].z, xr.w + sr.w + br4[q].w};
      const float zz[4] = {xz.x + sz.x + bz4[q].x, xz.y + sz.y + bz4[q].y,
                           xz.z + sz.z + bz4[q].z, xz.w + sz.w + bz4[q].w};
      const float gn[4] = {sn.x + bn4[q].x, sn.y + bn4[q].y,
                           sn.z + bn4[q].z, sn.w + bn4[q].w};
      const float xx[4] = {xn.x, xn.y, xn.z, xn.w};
      const float hh4[4] = {hc.x, hc.y, hc.z, hc.w};
#pragma unroll
      for (int e = 0; e < 4; ++e) {
        float r = 1.f / (1.f + __expf(-rr[e]));
        float z = 1.f / (1.f + __expf(-zz[e]));
        float na = xx[e] + r * gn[e];
        float ex = __expf(-2.f * na);
        float n = (1.f - ex) / (1.f + ex);
        hn4[e] = (1.f - z) * n + z * hh4[e];
      }
      float4 hv = make_float4(hn4[0], hn4[1], hn4[2], hn4[3]);
      ((float4*)(hsh + row * Hn + c))[0] = hv;
      if (store_seq)
        ((float4*)(out_seq + (unsigned)(((b0 + row) * Tn + t) * Hn + c)))[0] = hv;
      else if (t == Tn - 1)
        ((float4*)(out_last + (unsigned)((b0 + row) * Hn + c)))[0] = hv;
      // scatter packed bf16 pairs into next step's A-fragment buffer
#pragma unroll
      for (int p = 0; p < 2; ++p) {
        int cp = c + 2 * p;
        int kt = cp >> 5, kloc = cp & 31;
        int hh, jj;
        if (kloc < 16) { hh = kloc >> 3; jj = (kloc >> 1) & 3; }
        else           { hh = (kloc - 16) >> 3; jj = 4 + ((kloc >> 1) & 3); }
        aswz[kt][row + 16 * hh][jj] = pack2(hn4[2 * p], hn4[2 * p + 1]);
      }
    }
    __syncthreads();
  }
}

// ---------------------------------------------------------------------------
// MLP head: out[b] = W2 * relu(W1 * h_last[b] + b1) + b2   (tiny, VALU)
// ---------------------------------------------------------------------------
__global__ __launch_bounds__(128) void head_kernel(
    const float* __restrict__ last, const float* __restrict__ W1,
    const float* __restrict__ b1, const float* __restrict__ W2,
    const float* __restrict__ b2, float* __restrict__ out) {
  __shared__ float lsh[Hn];
  __shared__ float red[128];
  int b = blockIdx.x, j = threadIdx.x;
  lsh[j]       = last[b * Hn + j];
  lsh[j + 128] = last[b * Hn + j + 128];
  __syncthreads();
  float d = b1[j];
  const float* wr = W1 + j * Hn;
  for (int k = 0; k < Hn; ++k) d += lsh[k] * wr[k];
  red[j] = fmaxf(d, 0.f) * W2[j];
  __syncthreads();
  for (int s = 64; s > 0; s >>= 1) {
    if (j < s) red[j] += red[j + s];
    __syncthreads();
  }
  if (j == 0) out[b] = red[0] + b2[0];
}

// ---------------------------------------------------------------------------
extern "C" void kernel_launch(void* const* d_in, const int* in_sizes, int n_in,
                              void* d_out, int out_size, void* d_ws,
                              size_t ws_size, hipStream_t stream) {
  (void)in_sizes; (void)n_in; (void)out_size; (void)ws_size;
  const float* x     = (const float*)d_in[0];
  const float* W_ih0 = (const float*)d_in[1];
  const float* W_hh0 = (const float*)d_in[2];
  const float* b_ih0 = (const float*)d_in[3];
  const float* b_hh0 = (const float*)d_in[4];
  const float* W_ih1 = (const float*)d_in[5];
  const float* W_hh1 = (const float*)d_in[6];
  const float* b_ih1 = (const float*)d_in[7];
  const float* b_hh1 = (const float*)d_in[8];
  const float* W1    = (const float*)d_in[9];
  const float* b1    = (const float*)d_in[10];
  const float* W2    = (const float*)d_in[11];
  const float* b2    = (const float*)d_in[12];
  float* out = (float*)d_out;

  char* ws = (char*)d_ws;
  size_t off = 0;
  auto carve = [&](size_t bytes) {
    void* p = ws + off;
    off = (off + bytes + 255) & ~(size_t)255;
    return p;
  };
  const size_t M = (size_t)Bsz * Tn;                 // 65536
  float*    gx      = (float*)   carve(M * G3 * sizeof(float));      // 201 MB
  float*    h1seq   = (float*)   carve(M * Hn * sizeof(float));      //  67 MB
  float*    h2last  = (float*)   carve((size_t)Bsz * Hn * sizeof(float));
  unsigned* Wih0sw  = (unsigned*)carve((size_t)G3 * In / 2 * sizeof(unsigned));
  unsigned* Whh0sw  = (unsigned*)carve((size_t)G3 * Hn / 2 * sizeof(unsigned));
  unsigned* Wih1sw  = (unsigned*)carve((size_t)G3 * Hn / 2 * sizeof(unsigned));
  unsigned* Whh1sw  = (unsigned*)carve((size_t)G3 * Hn / 2 * sizeof(unsigned));

  swizzle_weights<<<(G3 * In / 2 + 255) / 256, 256, 0, stream>>>(W_ih0, Wih0sw, G3, In);
  swizzle_weights<<<(G3 * Hn / 2 + 255) / 256, 256, 0, stream>>>(W_hh0, Whh0sw, G3, Hn);
  swizzle_weights<<<(G3 * Hn / 2 + 255) / 256, 256, 0, stream>>>(W_ih1, Wih1sw, G3, Hn);
  swizzle_weights<<<(G3 * Hn / 2 + 255) / 256, 256, 0, stream>>>(W_hh1, Whh1sw, G3, Hn);

  const int gemm_blocks = (int)(M / 16);             // 4096 (one M-tile/block)

  gemm_bf16_wmma<<<gemm_blocks, 256, 0, stream>>>(x, Wih0sw, b_ih0, gx, In);
  gru_recurrent<<<Bsz / 16, 256, 0, stream>>>(gx, Whh0sw, b_hh0,
                                              h1seq, h2last, 1);

  gemm_bf16_wmma<<<gemm_blocks, 256, 0, stream>>>(h1seq, Wih1sw, b_ih1, gx, Hn);
  gru_recurrent<<<Bsz / 16, 256, 0, stream>>>(gx, Whh1sw, b_hh1,
                                              h1seq /*unused*/, h2last, 0);

  head_kernel<<<Bsz, 128, 0, stream>>>(h2last, W1, b1, W2, b2, out);
}
